// VuIjcai_249108103234
// MI455X (gfx1250) — matrix operand order repaired and checked by reading, not compiled, not run on hardware
//
#include <hip/hip_runtime.h>
#include <math.h>

// ---------------------------------------------------------------------------
// MI455X (gfx1250) attention-pooling classifier head.
//   Z = tanh(X @ W^T + b)                       (bf16 WMMA GEMM, tanh epilogue)
//   logits[b,l] = sum_s softmax_s(U Z^T)[l,s] * (F X^T)[l,s] + fb[l]
// Flash-attention-style streaming over S with online softmax; the [B,L,S]
// score tensor and [B,L,D] hidden tensor are never materialized.
// bf16 working set (~102MB) fits in the 192MB L2. The attention kernel
// stages the shared Z/X s-chunks into LDS with GLOBAL_LOAD_ASYNC_TO_LDS_B128
// (double buffered, ASYNCcnt-tracked) so all 8 waves of the WG read B-frags
// from LDS instead of 8x-redundant global loads.
// ---------------------------------------------------------------------------

typedef __attribute__((ext_vector_type(8)))  __bf16 v8bf;
typedef __attribute__((ext_vector_type(16))) __bf16 v16bf;
typedef __attribute__((ext_vector_type(8)))  float  v8f;

#define NB 4
#define NS 4096
#define ND 1024
#define NL 8192

// ---- f32 -> bf16 (round to nearest even), grid-stride --------------------
__global__ void __launch_bounds__(256) cvt_bf16_kernel(const float* __restrict__ in,
                                                       unsigned short* __restrict__ out,
                                                       long long n) {
    long long i = (long long)blockIdx.x * blockDim.x + threadIdx.x;
    long long stride = (long long)gridDim.x * blockDim.x;
    for (; i < n; i += stride) {
        unsigned int u = __float_as_uint(in[i]);
        unsigned int r = u + 0x7FFFu + ((u >> 16) & 1u);
        out[i] = (unsigned short)(r >> 16);
    }
}

// ---- WMMA fragment loaders (ISA 7.12.2 layouts, bf16 16x16x32) -----------
// A-matrix 16x32 (MxK): lane l<16 holds row l, K = {k0..k0+7, k0+16..k0+23};
//                       lane l+16 holds row l, K = {k0+8..15, k0+24..31}.
__device__ __forceinline__ v16bf load_a_frag(const unsigned short* __restrict__ rowp,
                                             int k0, int half) {
    v8bf lo = *reinterpret_cast<const v8bf*>(rowp + k0 + half * 8);
    v8bf hi = *reinterpret_cast<const v8bf*>(rowp + k0 + 16 + half * 8);
    return __builtin_shufflevector(lo, hi, 0, 1, 2, 3, 4, 5, 6, 7,
                                           8, 9, 10, 11, 12, 13, 14, 15);
}
// B-matrix 32x16 (KxN): lane n<16 holds column n (row n of the row-major
// second operand of A*B^T), K = {k0..k0+7, k0+16..k0+23}; lane n+16 holds K = {k0+8..15, k0+24..31}.
__device__ __forceinline__ v16bf load_b_frag(const unsigned short* rowp,
                                             int k0, int half) {
    v8bf lo = *reinterpret_cast<const v8bf*>(rowp + k0 + half * 8);
    v8bf hi = *reinterpret_cast<const v8bf*>(rowp + k0 + 16 + half * 8);
    return __builtin_shufflevector(lo, hi, 0, 1, 2, 3, 4, 5, 6, 7,
                                           8, 9, 10, 11, 12, 13, 14, 15);
}

// ---- CDNA5 async copy: 16 bytes/lane global -> LDS (ASYNCcnt) ------------
__device__ __forceinline__ void async_copy_b128(unsigned lds_byte_addr,
                                                const void* gaddr) {
    asm volatile("global_load_async_to_lds_b128 %0, %1, off"
                 :: "v"(lds_byte_addr), "v"(gaddr)
                 : "memory");
}
__device__ __forceinline__ void wait_asynccnt0() {
    asm volatile("s_wait_asynccnt 0" ::: "memory");
}
// Stage one contiguous 32KB block (16 rows x 1024 bf16) into LDS; 256 threads
// x 16B x 8 passes. Issued asynchronously; caller waits on ASYNCcnt.
__device__ __forceinline__ void stage32k(unsigned short* dst,
                                         const unsigned short* src, int tid) {
#pragma unroll
    for (int i = 0; i < 8; ++i) {
        unsigned boff = (unsigned)(i * 4096 + tid * 16);
        unsigned lds  = (unsigned)(size_t)((char*)dst + boff);
        async_copy_b128(lds, (const char*)src + boff);
    }
}

// ---- Stage 2: Z = tanh(X @ W^T + b), bf16 out ----------------------------
// Each wave computes a 16x64 strip (4 N-tiles) so every A-fragment is reused
// by 4 WMMAs. X: [M=NB*NS, ND] bf16; W: [ND, ND] bf16 (rows = e).
__global__ void __launch_bounds__(256) z_gemm_tanh_kernel(
        const unsigned short* __restrict__ X,
        const unsigned short* __restrict__ W,
        const float* __restrict__ Wb,
        unsigned short* __restrict__ Z) {
    const int ngroups_n = ND / 64;      // 16
    int wid  = blockIdx.x * (blockDim.x >> 5) + (threadIdx.x >> 5);
    int mt   = wid / ngroups_n;
    int ng   = wid % ngroups_n;
    int m0   = mt * 16;
    int n0   = ng * 64;

    int lane = threadIdx.x & 31;
    int lr   = lane & 15;
    int half = lane >> 4;

    const unsigned short* arow = X + (size_t)(m0 + lr) * ND;
    const unsigned short* brow0 = W + (size_t)(n0 + lr) * ND;

    v8f acc[4] = {{}, {}, {}, {}};
#pragma unroll 2
    for (int k0 = 0; k0 < ND; k0 += 32) {
        v16bf a = load_a_frag(arow, k0, half);
#pragma unroll
        for (int j = 0; j < 4; ++j) {
            v16bf b = load_b_frag(brow0 + (size_t)(16 * j) * ND, k0, half);
            acc[j] = __builtin_amdgcn_wmma_f32_16x16x32_bf16(
                false, a, false, b, (short)0, acc[j], false, false);
        }
    }

    // C/D layout: half h, slot r -> (row m0+8h+r, col n0+16j+lr)
#pragma unroll
    for (int j = 0; j < 4; ++j) {
        int col = n0 + 16 * j + lr;
        float bias = Wb[col];
#pragma unroll
        for (int r = 0; r < 8; ++r) {
            int row = m0 + 8 * half + r;
            float t = tanhf(acc[j][r] + bias);
            unsigned int u  = __float_as_uint(t);
            unsigned int rr = u + 0x7FFFu + ((u >> 16) & 1u);
            Z[(size_t)row * ND + col] = (unsigned short)(rr >> 16);
        }
    }
}

// ---- Stage 3: fused scores/softmax/G/logits (flash style) ----------------
// WG = 8 waves = 128 L-rows, one batch. Z/X s-chunks (16 rows each) are
// shared by all waves: double-buffered async LDS staging. Per chunk each
// wave runs 2x32 bf16 WMMAs (scores + G) then an online-softmax update with
// wave32 butterfly reductions over the 16-lane half.
__global__ void __launch_bounds__(256) attn_logits_kernel(
        const unsigned short* __restrict__ U,
        const unsigned short* __restrict__ F,
        const unsigned short* __restrict__ Z,
        const unsigned short* __restrict__ X,
        const float* __restrict__ fb,
        float* __restrict__ out) {
    extern __shared__ unsigned short smem[];   // 2 buffers x (Z 16K + X 16K) elems

    int tid  = threadIdx.x;
    int wave = tid >> 5;
    int lane = tid & 31;
    int lr   = lane & 15;
    int half = lane >> 4;

    int l0 = blockIdx.x * 128 + wave * 16;
    int b  = blockIdx.y;

    const unsigned short* urow = U + (size_t)(l0 + lr) * ND;
    const unsigned short* frow = F + (size_t)(l0 + lr) * ND;
    const unsigned short* Zb   = Z + (size_t)b * NS * ND;
    const unsigned short* Xb   = X + (size_t)b * NS * ND;

    float mrow[8], lrow[8], orow[8];
#pragma unroll
    for (int r = 0; r < 8; ++r) { mrow[r] = -3.0e38f; lrow[r] = 0.f; orow[r] = 0.f; }

    // Prologue: stage chunk 0 into buffer 0.
    stage32k(&smem[0],     Zb, tid);
    stage32k(&smem[16384], Xb, tid);
    wait_asynccnt0();
    __syncthreads();

    int cur = 0;
    for (int s0 = 0; s0 < NS; s0 += 16) {
        int nxt = cur ^ 1;
        if (s0 + 16 < NS) {   // async-prefetch next chunk while computing
            stage32k(&smem[nxt * 32768],         Zb + (size_t)(s0 + 16) * ND, tid);
            stage32k(&smem[nxt * 32768 + 16384], Xb + (size_t)(s0 + 16) * ND, tid);
        }
        const unsigned short* zls = &smem[cur * 32768];
        const unsigned short* xls = &smem[cur * 32768 + 16384];
        const unsigned short* zrow = zls + (size_t)lr * ND;
        const unsigned short* xrow = xls + (size_t)lr * ND;

        v8f accS = {}, accG = {};
#pragma unroll 2
        for (int k0 = 0; k0 < ND; k0 += 32) {
            v16bf au = load_a_frag(urow, k0, half);
            v16bf bz = load_b_frag(zrow, k0, half);
            accS = __builtin_amdgcn_wmma_f32_16x16x32_bf16(
                false, au, false, bz, (short)0, accS, false, false);
            v16bf af = load_a_frag(frow, k0, half);
            v16bf bx = load_b_frag(xrow, k0, half);
            accG = __builtin_amdgcn_wmma_f32_16x16x32_bf16(
                false, af, false, bx, (short)0, accG, false, false);
        }

        // Online softmax; row r of this half lives across lanes
        // [16*half .. 16*half+15]; xor offsets <16 stay inside the half.
#pragma unroll
        for (int r = 0; r < 8; ++r) {
            float sv = accS[r];
            float gv = accG[r];
            float rm = sv;
            rm = fmaxf(rm, __shfl_xor(rm, 8, 32));
            rm = fmaxf(rm, __shfl_xor(rm, 4, 32));
            rm = fmaxf(rm, __shfl_xor(rm, 2, 32));
            rm = fmaxf(rm, __shfl_xor(rm, 1, 32));
            float mnew = fmaxf(mrow[r], rm);
            float corr = __expf(mrow[r] - mnew);
            float p    = __expf(sv - mnew);
            float ps = p;
            float pg = p * gv;
            ps += __shfl_xor(ps, 8, 32);  pg += __shfl_xor(pg, 8, 32);
            ps += __shfl_xor(ps, 4, 32);  pg += __shfl_xor(pg, 4, 32);
            ps += __shfl_xor(ps, 2, 32);  pg += __shfl_xor(pg, 2, 32);
            ps += __shfl_xor(ps, 1, 32);  pg += __shfl_xor(pg, 1, 32);
            lrow[r] = lrow[r] * corr + ps;
            orow[r] = orow[r] * corr + pg;
            mrow[r] = mnew;
        }

        // Next chunk landed + everyone done reading cur before reuse.
        wait_asynccnt0();
        __syncthreads();
        cur = nxt;
    }

    if (lr == 0) {
#pragma unroll
        for (int r = 0; r < 8; ++r) {
            int row = l0 + 8 * half + r;
            out[(size_t)b * NL + row] = orow[r] / lrow[r] + fb[row];
        }
    }
}

// ---------------------------------------------------------------------------
extern "C" void kernel_launch(void* const* d_in, const int* in_sizes, int n_in,
                              void* d_out, int out_size, void* d_ws, size_t ws_size,
                              hipStream_t stream) {
    const float* inputs  = (const float*)d_in[0];   // [B,S,D]
    const float* W_w     = (const float*)d_in[1];   // [D,D]
    const float* W_b     = (const float*)d_in[2];   // [D]
    const float* U_w     = (const float*)d_in[3];   // [L,D]
    const float* final_w = (const float*)d_in[4];   // [L,D]
    const float* final_b = (const float*)d_in[5];   // [L]
    float* out = (float*)d_out;                     // [B,L]

    const size_t nX = (size_t)NB * NS * ND;   // 16.78M
    const size_t nW = (size_t)ND * ND;        // 1.05M
    const size_t nU = (size_t)NL * ND;        // 8.39M

    char* ws = (char*)d_ws;
    unsigned short* Xb  = (unsigned short*)ws; ws += nX * 2;
    unsigned short* Wb  = (unsigned short*)ws; ws += nW * 2;
    unsigned short* Ub  = (unsigned short*)ws; ws += nU * 2;
    unsigned short* Fb  = (unsigned short*)ws; ws += nU * 2;
    unsigned short* Zb  = (unsigned short*)ws; ws += nX * 2;  // ~102.7MB total

    cvt_bf16_kernel<<<2048, 256, 0, stream>>>(inputs,  Xb, (long long)nX);
    cvt_bf16_kernel<<<512,  256, 0, stream>>>(W_w,     Wb, (long long)nW);
    cvt_bf16_kernel<<<1024, 256, 0, stream>>>(U_w,     Ub, (long long)nU);
    cvt_bf16_kernel<<<1024, 256, 0, stream>>>(final_w, Fb, (long long)nU);

    // Z GEMM: (NB*NS/16) M-tiles x (ND/64) N-groups = 16384 waves, 8/block
    z_gemm_tanh_kernel<<<16384 / 8, 256, 0, stream>>>(Xb, Wb, W_b, Zb);

    // Fused attention: grid = (L/128, B), 128KB dynamic LDS (double buffer)
    attn_logits_kernel<<<dim3(NL / 128, NB), 256, 131072, stream>>>(
        Ub, Fb, Zb, Xb, final_b, out);
}
